// TinyTransformerLayer_19447611916837
// MI455X (gfx1250) — compile-verified, matrix operand order checked by reading, not run on hardware
//
#include <hip/hip_runtime.h>
#include <hip/hip_bf16.h>
#include <math.h>

typedef __bf16 bf16;
typedef __attribute__((ext_vector_type(16))) __bf16 v16bf;
typedef __attribute__((ext_vector_type(8)))  __bf16 v8bf;
typedef __attribute__((ext_vector_type(8)))  float  v8f;
typedef __attribute__((ext_vector_type(4)))  float  v4f;

#define DIMN   1024
#define S_LEN  2048
#define BATCH  2
#define NHEAD  16
#define HDIM   64
#define HIDDEN 4096
#define MTOT   (BATCH * S_LEN)   // 4096 rows

union Frag { v16bf v; v8bf h[2]; };

static __device__ __forceinline__ v8f wmma_bf16(const Frag& a, const Frag& b, v8f c) {
    return __builtin_amdgcn_wmma_f32_16x16x32_bf16(false, a.v, false, b.v,
                                                   (short)0, c, false, false);
}

// XOR-shuffle within each 16-lane half using v_permlane16_b32 (pure VALU,
// avoids ds_bpermute + s_wait_dscnt round-trips in the flash-attention loop).
template <int M>
static __device__ __forceinline__ float permxor16(float v) {
#if __has_builtin(__builtin_amdgcn_permlane16)
    constexpr unsigned lo =
        ((unsigned)(0 ^ M)) | ((unsigned)(1 ^ M) << 4) | ((unsigned)(2 ^ M) << 8) |
        ((unsigned)(3 ^ M) << 12) | ((unsigned)(4 ^ M) << 16) |
        ((unsigned)(5 ^ M) << 20) | ((unsigned)(6 ^ M) << 24) |
        ((unsigned)(7 ^ M) << 28);
    constexpr unsigned hi =
        ((unsigned)((8 ^ M) & 15)) | ((unsigned)((9 ^ M) & 15) << 4) |
        ((unsigned)((10 ^ M) & 15) << 8) | ((unsigned)((11 ^ M) & 15) << 12) |
        ((unsigned)((12 ^ M) & 15) << 16) | ((unsigned)((13 ^ M) & 15) << 20) |
        ((unsigned)((14 ^ M) & 15) << 24) | ((unsigned)((15 ^ M) & 15) << 28);
    int iv = __builtin_bit_cast(int, v);
    int r = __builtin_amdgcn_permlane16(iv, iv, (int)lo, (int)hi, false, false);
    return __builtin_bit_cast(float, r);
#else
    return __shfl_xor(v, M, 32);
#endif
}

static __device__ __forceinline__ float hmax16(float v) {
    v = fmaxf(v, permxor16<1>(v));
    v = fmaxf(v, permxor16<2>(v));
    v = fmaxf(v, permxor16<4>(v));
    v = fmaxf(v, permxor16<8>(v));
    return v;
}
static __device__ __forceinline__ float hsum16(float v) {
    v += permxor16<1>(v);
    v += permxor16<2>(v);
    v += permxor16<4>(v);
    v += permxor16<8>(v);
    return v;
}

// Async global->LDS 16B copy (CDNA5 GLOBAL_LOAD_ASYNC_TO_LDS_B128, ASYNCcnt).
static __device__ __forceinline__ void async_copy_b128(unsigned lds_off,
                                                       const void* gaddr) {
    asm volatile("global_load_async_to_lds_b128 %0, %1, off"
                 :: "v"(lds_off), "v"((unsigned long long)(size_t)gaddr)
                 : "memory");
}

// ---------------- fp32 -> bf16 convert (weights) ----------------
__global__ __launch_bounds__(256) void cvt_kernel(const float* __restrict__ src,
                                                  bf16* __restrict__ dst, int n) {
    int i = (blockIdx.x * 256 + threadIdx.x) * 4;
    if (i < n) {
        v4f s = *(const v4f*)(src + i);
#pragma unroll
        for (int t = 0; t < 4; ++t) dst[i + t] = (bf16)s[t];
    }
}

// ---------------- RMSNorm: fp32 row -> bf16 row ----------------
__global__ __launch_bounds__(256) void rmsnorm_kernel(const float* __restrict__ x,
                                                      const float* __restrict__ w,
                                                      bf16* __restrict__ out) {
    __shared__ float red[8];
    const int row = blockIdx.x;
    const int tid = threadIdx.x;
    const float* xr = x + (size_t)row * DIMN;
    v4f xv = *(const v4f*)(xr + tid * 4);
    float ss = xv.x * xv.x + xv.y * xv.y + xv.z * xv.z + xv.w * xv.w;
    ss += permxor16<1>(ss);
    ss += permxor16<2>(ss);
    ss += permxor16<4>(ss);
    ss += permxor16<8>(ss);
    ss += __shfl_xor(ss, 16, 32);
    if ((tid & 31) == 0) red[tid >> 5] = ss;
    __syncthreads();
    float tot = 0.f;
#pragma unroll
    for (int i = 0; i < 8; ++i) tot += red[i];
    const float rinv = rsqrtf(tot * (1.f / DIMN) + 1e-5f);
    const float* wr = w + tid * 4;
    bf16* orow = out + (size_t)row * DIMN + tid * 4;
#pragma unroll
    for (int t = 0; t < 4; ++t) orow[t] = (bf16)(xv[t] * rinv * wr[t]);
}

// ---------------- WMMA GEMM: C[M,N] = A[M,K] @ W[N,K]^T ----------------
// Double-buffered LDS staging via async global->LDS DMA (ASYNCcnt).
// MODE 0: QKV scatter (N=3072): q/k -> [B,H,S,Dh] bf16, v -> [B,H,Dh,S] bf16
// MODE 1: store bf16 C[M,N]
// MODE 2: store fp32 C[M,N] + res[M,N]
template <int MODE>
__global__ __launch_bounds__(256)
void gemm_kernel(const bf16* __restrict__ A, const bf16* __restrict__ W,
                 int M, int N, int K,
                 float* __restrict__ outF, bf16* __restrict__ outB,
                 const float* __restrict__ res,
                 bf16* __restrict__ qo, bf16* __restrict__ ko, bf16* __restrict__ vo) {
    __shared__ bf16 As[2][128 * 40];
    __shared__ bf16 Bs[2][128 * 40];
    const int m0 = blockIdx.y * 128;
    const int n0 = blockIdx.x * 128;
    const int tid = threadIdx.x;
    const int lane = tid & 31;
    const int wave = tid >> 5;
    const int wm = (wave >> 1) * 32;   // wave M offset (4 waves in M)
    const int wn = (wave & 1) * 64;    // wave N offset (2 waves in N)
    const int lo = lane & 15;
    const int hi = lane >> 4;
    const int kb = hi << 3;            // K-chunk base per A/B VGPR layout

    v8f acc[2][4] = {};

    // Each thread stages 2 chunks of A + 2 chunks of B = 4 async b128 ops/tile.
    const int row0 = tid >> 2, col0 = (tid & 3) << 3;        // chunk c = tid
    const int row1 = (tid + 256) >> 2, col1 = col0;          // chunk c = tid+256

    auto stage = [&](int k0, int pb) {
        async_copy_b128((unsigned)(size_t)&As[pb][row0 * 40 + col0],
                        &A[(size_t)(m0 + row0) * K + k0 + col0]);
        async_copy_b128((unsigned)(size_t)&Bs[pb][row0 * 40 + col0],
                        &W[(size_t)(n0 + row0) * K + k0 + col0]);
        async_copy_b128((unsigned)(size_t)&As[pb][row1 * 40 + col1],
                        &A[(size_t)(m0 + row1) * K + k0 + col1]);
        async_copy_b128((unsigned)(size_t)&Bs[pb][row1 * 40 + col1],
                        &W[(size_t)(n0 + row1) * K + k0 + col1]);
    };

    unsigned p = 0;
    stage(0, 0);
    for (int k0 = 0; k0 < K; k0 += 32) {
        if (k0 + 32 < K) {
            stage(k0 + 32, p ^ 1);
            // previous tile's 4 ops (per thread) complete when count <= 4
            asm volatile("s_wait_asynccnt 0x4" ::: "memory");
        } else {
            asm volatile("s_wait_asynccnt 0x0" ::: "memory");
        }
        __syncthreads();
        const bf16* Asp = As[p];
        const bf16* Bsp = Bs[p];
        Frag a[2], b;
#pragma unroll
        for (int mt = 0; mt < 2; ++mt) {
            const bf16* ap = &Asp[(wm + mt * 16 + lo) * 40 + kb];
            a[mt].h[0] = *(const v8bf*)ap;
            a[mt].h[1] = *(const v8bf*)(ap + 16);
        }
#pragma unroll
        for (int nt = 0; nt < 4; ++nt) {
            const bf16* bp = &Bsp[(wn + nt * 16 + lo) * 40 + kb];
            b.h[0] = *(const v8bf*)bp;
            b.h[1] = *(const v8bf*)(bp + 16);
            acc[0][nt] = wmma_bf16(a[0], b, acc[0][nt]);
            acc[1][nt] = wmma_bf16(a[1], b, acc[1][nt]);
        }
        __syncthreads();   // protect buffer p from next iteration's DMA
        p ^= 1;
    }

#pragma unroll
    for (int mt = 0; mt < 2; ++mt)
#pragma unroll
        for (int nt = 0; nt < 4; ++nt)
#pragma unroll
            for (int r = 0; r < 8; ++r) {
                const int m = m0 + wm + mt * 16 + r + hi * 8;
                const int n = n0 + wn + nt * 16 + lo;
                const float vacc = acc[mt][nt][r];
                if constexpr (MODE == 0) {
                    const int bq = m >> 11, s = m & 2047;
                    const int which = n >> 10, col = n & 1023;
                    const int hh = col >> 6, d = col & 63;
                    const size_t bhd = (size_t)(bq * NHEAD + hh);
                    if (which == 0)
                        qo[(bhd * S_LEN + s) * HDIM + d] = (bf16)vacc;
                    else if (which == 1)
                        ko[(bhd * S_LEN + s) * HDIM + d] = (bf16)vacc;
                    else
                        vo[(bhd * HDIM + d) * S_LEN + s] = (bf16)vacc;
                } else if constexpr (MODE == 1) {
                    outB[(size_t)m * N + n] = (bf16)vacc;
                } else {
                    outF[(size_t)m * N + n] = vacc + res[(size_t)m * N + n];
                }
            }
}

// ---------------- Flash attention (causal) via WMMA ----------------
// q,k: [B*H, S, 64] bf16 ; v: [B*H, 64, S] bf16 ; out: [B, S, 1024] bf16
__global__ __launch_bounds__(128)
void attn_kernel(const bf16* __restrict__ q, const bf16* __restrict__ k,
                 const bf16* __restrict__ v, bf16* __restrict__ out) {
    __shared__ bf16 plds[4][16 * 40];
    const int tid = threadIdx.x;
    const int lane = tid & 31;
    const int wave = tid >> 5;
    const int lo = lane & 15;
    const int hi = lane >> 4;
    const int kb = hi << 3;
    const int bh = blockIdx.y;                   // 0..31
    const int qbase = blockIdx.x * 64 + wave * 16;
    const int b = bh >> 4, h0 = bh & 15;

    const bf16* qp = q + (((size_t)bh) * S_LEN + qbase + lo) * HDIM;
    Frag qa[2];
    qa[0].h[0] = *(const v8bf*)(qp + kb);
    qa[0].h[1] = *(const v8bf*)(qp + kb + 16);
    qa[1].h[0] = *(const v8bf*)(qp + 32 + kb);
    qa[1].h[1] = *(const v8bf*)(qp + 32 + kb + 16);

    v8f o[4] = {};
    float mrow[8], lrow[8];
#pragma unroll
    for (int r = 0; r < 8; ++r) { mrow[r] = -INFINITY; lrow[r] = 0.f; }
    const float scale = 0.125f;  // 1/sqrt(64)
    bf16* pl = plds[wave];

    for (int j0 = 0; j0 <= qbase + 15; j0 += 32) {
        v8f s0 = {}, s1 = {};
        {
            const bf16* kp0 = k + (((size_t)bh) * S_LEN + j0 + lo) * HDIM;
            const bf16* kp1 = kp0 + (size_t)16 * HDIM;
            Frag b00, b01, b10, b11;
            b00.h[0] = *(const v8bf*)(kp0 + kb);
            b00.h[1] = *(const v8bf*)(kp0 + kb + 16);
            b01.h[0] = *(const v8bf*)(kp0 + 32 + kb);
            b01.h[1] = *(const v8bf*)(kp0 + 32 + kb + 16);
            b10.h[0] = *(const v8bf*)(kp1 + kb);
            b10.h[1] = *(const v8bf*)(kp1 + kb + 16);
            b11.h[0] = *(const v8bf*)(kp1 + 32 + kb);
            b11.h[1] = *(const v8bf*)(kp1 + 32 + kb + 16);
            s0 = wmma_bf16(qa[0], b00, s0);
            s0 = wmma_bf16(qa[1], b01, s0);
            s1 = wmma_bf16(qa[0], b10, s1);
            s1 = wmma_bf16(qa[1], b11, s1);
        }
        // Online softmax. Row statistics align with the C-matrix VGPR layout
        // (row = hi*8 + r, 16 columns spread across the 16-lane half).
        float alpha[8];
#pragma unroll
        for (int r = 0; r < 8; ++r) {
            const int m = qbase + hi * 8 + r;
            const float x0 = (j0 + lo      <= m) ? s0[r] * scale : -INFINITY;
            const float x1 = (j0 + 16 + lo <= m) ? s1[r] * scale : -INFINITY;
            const float mx = hmax16(fmaxf(x0, x1));
            const float mnew = fmaxf(mrow[r], mx);
            alpha[r] = __expf(mrow[r] - mnew);
            const float p0 = __expf(x0 - mnew);
            const float p1 = __expf(x1 - mnew);
            lrow[r] = lrow[r] * alpha[r] + hsum16(p0 + p1);
            mrow[r] = mnew;
            pl[(hi * 8 + r) * 40 + lo]      = (bf16)p0;
            pl[(hi * 8 + r) * 40 + 16 + lo] = (bf16)p1;
        }
#pragma unroll
        for (int dt = 0; dt < 4; ++dt) {
            v8f ov = o[dt];
#pragma unroll
            for (int r = 0; r < 8; ++r) ov[r] *= alpha[r];
            o[dt] = ov;
        }
        asm volatile("s_wait_dscnt 0x0" ::: "memory");
        Frag pa;
        pa.h[0] = *(const v8bf*)&pl[lo * 40 + kb];
        pa.h[1] = *(const v8bf*)&pl[lo * 40 + kb + 16];
#pragma unroll
        for (int dt = 0; dt < 4; ++dt) {
            const bf16* vp = v + (((size_t)bh) * HDIM + dt * 16 + lo) * S_LEN + j0;
            Frag vb;
            vb.h[0] = *(const v8bf*)(vp + kb);
            vb.h[1] = *(const v8bf*)(vp + kb + 16);
            o[dt] = wmma_bf16(pa, vb, o[dt]);
        }
        asm volatile("s_wait_dscnt 0x0" ::: "memory");
    }

#pragma unroll
    for (int dt = 0; dt < 4; ++dt)
#pragma unroll
        for (int r = 0; r < 8; ++r) {
            const int s = qbase + hi * 8 + r;
            const int d = dt * 16 + lo;
            out[((size_t)(b * S_LEN + s)) * DIMN + h0 * HDIM + d] =
                (bf16)(o[dt][r] / lrow[r]);
        }
}

// ---------------- silu(gate) * up -> bf16 ----------------
__global__ __launch_bounds__(256)
void silu_mul_kernel(const bf16* __restrict__ gu, bf16* __restrict__ outm) {
    const size_t idx = ((size_t)blockIdx.x * 256 + threadIdx.x) * 8;
    const size_t m = idx >> 12;          // / 4096
    const size_t j = idx & 4095;
    v8bf gv = *(const v8bf*)&gu[m * 8192 + j];
    v8bf uv = *(const v8bf*)&gu[m * 8192 + 4096 + j];
    v8bf ov;
#pragma unroll
    for (int t = 0; t < 8; ++t) {
        const float g = (float)gv[t];
        const float u = (float)uv[t];
        ov[t] = (bf16)((g / (1.f + __expf(-g))) * u);
    }
    *(v8bf*)&outm[idx] = ov;
}

extern "C" void kernel_launch(void* const* d_in, const int* in_sizes, int n_in,
                              void* d_out, int out_size, void* d_ws, size_t ws_size,
                              hipStream_t stream) {
    const float* x     = (const float*)d_in[0];
    const float* wq    = (const float*)d_in[1];
    const float* wk    = (const float*)d_in[2];
    const float* wv    = (const float*)d_in[3];
    const float* wo    = (const float*)d_in[4];
    const float* wgate = (const float*)d_in[5];
    const float* wup   = (const float*)d_in[6];
    const float* wdown = (const float*)d_in[7];
    const float* n1w   = (const float*)d_in[8];
    const float* n2w   = (const float*)d_in[9];

    char* ws = (char*)d_ws;
    size_t off = 0;
    auto take = [&](size_t bytes) -> char* {
        char* p = ws + off;
        off += (bytes + 255) & ~(size_t)255;
        return p;
    };
    bf16*  a1   = (bf16*)take((size_t)MTOT * DIMN * 2);
    bf16*  wqkv = (bf16*)take((size_t)3 * DIMN * DIMN * 2);
    bf16*  wo_b = (bf16*)take((size_t)DIMN * DIMN * 2);
    bf16*  wgu  = (bf16*)take((size_t)2 * HIDDEN * DIMN * 2);
    bf16*  wdn  = (bf16*)take((size_t)DIMN * HIDDEN * 2);
    bf16*  qws  = (bf16*)take((size_t)MTOT * DIMN * 2);
    bf16*  kws  = (bf16*)take((size_t)MTOT * DIMN * 2);
    bf16*  vws  = (bf16*)take((size_t)MTOT * DIMN * 2);
    bf16*  attn = (bf16*)take((size_t)MTOT * DIMN * 2);
    float* x2   = (float*)take((size_t)MTOT * DIMN * 4);
    bf16*  a2   = (bf16*)take((size_t)MTOT * DIMN * 2);
    bf16*  gu   = (bf16*)take((size_t)MTOT * 2 * HIDDEN * 2);
    bf16*  mlp  = (bf16*)take((size_t)MTOT * HIDDEN * 2);

    auto cvt = [&](const float* s, bf16* dp, int n) {
        cvt_kernel<<<(n / 4 + 255) / 256, 256, 0, stream>>>(s, dp, n);
    };
    cvt(wq, wqkv, DIMN * DIMN);
    cvt(wk, wqkv + (size_t)DIMN * DIMN, DIMN * DIMN);
    cvt(wv, wqkv + (size_t)2 * DIMN * DIMN, DIMN * DIMN);
    cvt(wo, wo_b, DIMN * DIMN);
    cvt(wgate, wgu, HIDDEN * DIMN);
    cvt(wup, wgu + (size_t)HIDDEN * DIMN, HIDDEN * DIMN);
    cvt(wdown, wdn, DIMN * HIDDEN);

    rmsnorm_kernel<<<MTOT, 256, 0, stream>>>(x, n1w, a1);

    gemm_kernel<0><<<dim3(3072 / 128, MTOT / 128), 256, 0, stream>>>(
        a1, wqkv, MTOT, 3072, DIMN, nullptr, nullptr, nullptr, qws, kws, vws);

    attn_kernel<<<dim3(S_LEN / 64, BATCH * NHEAD), 128, 0, stream>>>(qws, kws, vws, attn);

    gemm_kernel<2><<<dim3(DIMN / 128, MTOT / 128), 256, 0, stream>>>(
        attn, wo_b, MTOT, DIMN, DIMN, x2, nullptr, x, nullptr, nullptr, nullptr);

    rmsnorm_kernel<<<MTOT, 256, 0, stream>>>(x2, n2w, a2);

    gemm_kernel<1><<<dim3(2 * HIDDEN / 128, MTOT / 128), 256, 0, stream>>>(
        a2, wgu, MTOT, 2 * HIDDEN, DIMN, nullptr, gu, nullptr, nullptr, nullptr, nullptr);

    silu_mul_kernel<<<((size_t)MTOT * HIDDEN / 8) / 256, 256, 0, stream>>>(gu, mlp);

    gemm_kernel<2><<<dim3(DIMN / 128, MTOT / 128), 256, 0, stream>>>(
        mlp, wdn, MTOT, DIMN, HIDDEN, (float*)d_out, nullptr, x2,
        nullptr, nullptr, nullptr);
}
